// GridRnn_78030965833769
// MI455X (gfx1250) — matrix operand order repaired and checked by reading, not compiled, not run on hardware
//
#include <hip/hip_runtime.h>

typedef __bf16 bf16;
typedef __attribute__((ext_vector_type(16))) bf16 v16bf;
typedef __attribute__((ext_vector_type(8)))  float v8f;
typedef __attribute__((ext_vector_type(4)))  unsigned int u32x4;
typedef __attribute__((ext_vector_type(8)))  int i32x8;
typedef __attribute__((ext_vector_type(4)))  int i32x4;

constexpr int kE = 128, kH = 128, kL = 6, kC = 4, kO = 1024, kB = 16384;
constexpr int kTH = 3 * kH;   // 384

#define WMMA_BF16(a,b,c) __builtin_amdgcn_wmma_f32_16x16x32_bf16(false,(a),false,(b),(short)0,(c),false,false)

__device__ __forceinline__ float sigmoidf_(float x) { return 1.f / (1.f + __expf(-x)); }
__device__ __forceinline__ v8f zero8() { v8f z = {0.f,0.f,0.f,0.f,0.f,0.f,0.f,0.f}; return z; }

// A fragment (16x32 bf16): rows row0..row0+15 of row-major [*,stride] bf16 matrix, k in [k0,k0+32)
// ISA 7.12.2: lane m = lane&15; lanes16-31 offset K by +8; VGPR v holds K pair; v4..7 hold K+16.
__device__ __forceinline__ v16bf frag_a_bf(const bf16* __restrict__ base, int stride, int row0, int k0) {
  int lane = threadIdx.x & 31;
  const bf16* p = base + (size_t)(row0 + (lane & 15)) * stride + k0 + ((lane >> 4) << 3);
  v16bf f;
#pragma unroll
  for (int v = 0; v < 8; ++v) {
    int k = ((v >> 2) << 4) + ((v & 3) << 1);
    f[2*v]   = p[k];
    f[2*v+1] = p[k+1];
  }
  return f;
}

// Same, but source is fp32 (convert on the fly).
__device__ __forceinline__ v16bf frag_a_f32(const float* __restrict__ base, int stride, int row0, int k0) {
  int lane = threadIdx.x & 31;
  const float* p = base + (size_t)(row0 + (lane & 15)) * stride + k0 + ((lane >> 4) << 3);
  v16bf f;
#pragma unroll
  for (int v = 0; v < 8; ++v) {
    int k = ((v >> 2) << 4) + ((v & 3) << 1);
    f[2*v]   = (bf16)p[k];
    f[2*v+1] = (bf16)p[k+1];
  }
  return f;
}

// B fragment (32x16 bf16) for X @ W^T: B[k][n] = W[n0+n][k0+k], W row-major [N,stride].
// ISA: lane n = lane&15; lanes0-15 K=0..7,16..23, lanes16-31 K=8..15,24..31 (matches A).
__device__ __forceinline__ v16bf frag_b_w(const bf16* __restrict__ W, int stride, int n0, int k0) {
  int lane = threadIdx.x & 31;
  const bf16* p = W + (size_t)(n0 + (lane & 15)) * stride + k0 + ((lane >> 4) << 3);
  v16bf f;
#pragma unroll
  for (int v = 0; v < 16; ++v) f[v] = p[(v < 8) ? v : v + 8];
  return f;
}

// ---- Tensor Data Mover: DMA a [3 gates][16 rows][128 cols] bf16 weight slab into LDS ----
// D# per CDNA5 ISA: 3D tile, tile_dim0=128 (K elems), tile_dim1=16 (rows),
// tile_dim2=3 (gate blocks, stride 128 rows). data_size=1 (2 bytes). Written contiguously to LDS.
__device__ __forceinline__ void tdm_load_slab(unsigned lds_off, const bf16* gsrc) {
  unsigned long long ga = (unsigned long long)gsrc;
  u32x4 g0 = { 1u,                                   // count=1 valid descriptor
               lds_off,                              // lds_addr (bytes)
               (unsigned)(ga & 0xffffffffu),
               (unsigned)((ga >> 32) & 0x01ffffffu) | (2u << 30) };  // global_addr hi | type=2
  i32x8 g1 = { (1 << 16),        // wg_mask=0, data_size=1 (2B)
               (128 << 16),      // tensor_dim0 = 128 (bits 63:48)
               (384 << 16),      // tensor_dim0 hi=0 | tensor_dim1 = 384
               (128 << 16),      // tensor_dim1 hi=0 | tile_dim0 = 128
               16 | (3 << 16),   // tile_dim1 = 16, tile_dim2 = 3
               128,              // tensor_dim0_stride = 128
               (16384 << 16),    // stride0 hi=0 | tensor_dim1_stride = 16384 (lo16)
               0 };              // tensor_dim1_stride hi
  i32x4 g2 = { 3, 0, 0, 0 };     // tensor_dim2 = 3
  i32x4 g3 = { 0, 0, 0, 0 };
#if __clang_major__ >= 23
  i32x8 g4 = { 0,0,0,0,0,0,0,0 };
  __builtin_amdgcn_tensor_load_to_lds(g0, g1, g2, g3, g4, 0);
#else
  __builtin_amdgcn_tensor_load_to_lds(g0, g1, g2, g3, 0);
#endif
}

// ---------------- prep kernels ----------------
__global__ __launch_bounds__(256) void k_f2bf(const float* __restrict__ s, bf16* __restrict__ d, int n) {
  int i = blockIdx.x * 256 + threadIdx.x;
  if (i < n) d[i] = (bf16)s[i];
}

__global__ __launch_bounds__(128) void k_embed(const int* __restrict__ tokens, const float* __restrict__ emb,
                                               bf16* __restrict__ x0) {
  int b = blockIdx.x, e = threadIdx.x;
  int t = tokens[b];
  x0[(size_t)b * kE + e] = (bf16)emb[(size_t)t * kE + e];
}

// ---------------- fused GRU (gi + gh GEMMs + elementwise), TDM-staged weights ----------------
// grid (B/128, #cells), block 256. Writes hn in f32 and bf16.
// HAS_GI=false: layer-0 cells 1..3 where gi is bias-broadcast only (no x GEMM, no Wih DMA).
template <bool HAS_GI>
__global__ __launch_bounds__(256) void k_gru(
    const bf16* __restrict__ x,       // (C,B,H) bf16 (c_base applies), or (B,E) for layer 0 c=0
    const float* __restrict__ h,      // (C,B,H) f32 (slice of d_in h at layer l)
    const bf16* __restrict__ wih,     // (C,3H,H) bf16, or (3H,E) for layer 0 c=0
    const bf16* __restrict__ whh,     // (C,3H,H) bf16
    const float* __restrict__ bih,    // (C,3H)
    const float* __restrict__ bhh,    // (C,3H)
    float* __restrict__ hn_f32,       // (C,B,H)
    bf16* __restrict__ hn_bf,         // (C,B,H)
    int c_base) {
  // double-buffered weight slabs: [buf][mat: 0=wih,1=whh][gate][row 16][k 128]
  __shared__ bf16 wbuf[2][2][3][16][128];
  constexpr unsigned kMatBytes = 3 * 16 * 128 * 2;   // 12288
  constexpr unsigned kBufBytes = 2 * kMatBytes;      // 24576

  int c = blockIdx.y + c_base;
  int wave = threadIdx.x >> 5, lane = threadIdx.x & 31;
  int r0 = blockIdx.x * 128 + wave * 16;

  const bf16*  Xc  = x + (size_t)c * kB * kH;        // c==0 for layer 0
  const bf16*  Wih = wih + (size_t)c * kTH * kH;     // c==0 for layer 0
  const bf16*  Whh = whh + (size_t)c * kTH * kH;
  const float* Hc  = h + (size_t)c * kB * kH;

  v16bf xa[4], ha[4];
#pragma unroll
  for (int s = 0; s < 4; ++s) {
    ha[s] = frag_a_f32(Hc, kH, r0, 32 * s);
    if constexpr (HAS_GI) xa[s] = frag_a_bf(Xc, kH, r0, 32 * s);
  }

  // prologue: wave 0 DMAs the jt=0 slabs
  if (wave == 0) {
    if constexpr (HAS_GI) tdm_load_slab(0u, Wih);
    tdm_load_slab(kMatBytes, Whh);
  }

  // lambda: B fragment from a staged LDS slab (ds_load path, addrspace(3))
  auto ldsfrag = [&](int bsel, int mat, int q, int s) {
    v16bf f;
    int r = lane & 15;
    int kk = 32 * s + ((lane >> 4) << 3);
#pragma unroll
    for (int v = 0; v < 16; ++v) f[v] = wbuf[bsel][mat][q][r][kk + ((v < 8) ? v : v + 8)];
    return f;
  };

  for (int jt = 0; jt < 8; ++jt) {
    if (wave == 0) {
      if (jt < 7) {
        unsigned boff = (unsigned)((jt + 1) & 1) * kBufBytes;
        if constexpr (HAS_GI) tdm_load_slab(boff, Wih + (size_t)(jt + 1) * 16 * kH);
        tdm_load_slab(boff + kMatBytes, Whh + (size_t)(jt + 1) * 16 * kH);
        // in-order TENSORcnt: wait until only the just-issued next-buffer DMAs are outstanding
        if constexpr (HAS_GI) __builtin_amdgcn_s_wait_tensorcnt((short)2);
        else                  __builtin_amdgcn_s_wait_tensorcnt((short)1);
      } else {
        __builtin_amdgcn_s_wait_tensorcnt((short)0);
      }
    }
    __syncthreads();   // publish current slab to all waves

    int bsel = jt & 1;
    v8f ar = zero8(), az = zero8(), an = zero8();
    v8f br = zero8(), bz = zero8(), bn = zero8();
#pragma unroll
    for (int s = 0; s < 4; ++s) {
      v16bf w0 = ldsfrag(bsel, 1, 0, s);
      v16bf w1 = ldsfrag(bsel, 1, 1, s);
      v16bf w2 = ldsfrag(bsel, 1, 2, s);
      br = WMMA_BF16(ha[s], w0, br);
      bz = WMMA_BF16(ha[s], w1, bz);
      bn = WMMA_BF16(ha[s], w2, bn);
      if constexpr (HAS_GI) {
        v16bf g0 = ldsfrag(bsel, 0, 0, s);
        v16bf g1 = ldsfrag(bsel, 0, 1, s);
        v16bf g2 = ldsfrag(bsel, 0, 2, s);
        ar = WMMA_BF16(xa[s], g0, ar);
        az = WMMA_BF16(xa[s], g1, az);
        an = WMMA_BF16(xa[s], g2, an);
      }
    }
    int ncol = jt * 16 + (lane & 15);
    float bi_r = bih[c * kTH + ncol];
    float bi_z = bih[c * kTH + kH + ncol];
    float bi_n = bih[c * kTH + 2 * kH + ncol];
    float bh_r = bhh[c * kTH + ncol];
    float bh_z = bhh[c * kTH + kH + ncol];
    float bh_n = bhh[c * kTH + 2 * kH + ncol];
#pragma unroll
    for (int v = 0; v < 8; ++v) {
      int m = v + ((lane >> 4) << 3);
      int b = r0 + m;
      float r  = sigmoidf_(ar[v] + bi_r + br[v] + bh_r);
      float z  = sigmoidf_(az[v] + bi_z + bz[v] + bh_z);
      float nn = tanhf(an[v] + bi_n + r * (bn[v] + bh_n));
      float hold = Hc[(size_t)b * kH + ncol];
      float o = (1.f - z) * nn + z * hold;
      size_t idx = ((size_t)c * kB + b) * kH + ncol;
      hn_f32[idx] = o;
      hn_bf[idx]  = (bf16)o;
    }
    __syncthreads();   // all waves done with this slab before next TDM overwrites it
  }
}

// ---------------- generic GEMM: Y[M,N] = X_bf[M,128] @ W_bf[N,128]^T + bias ----------------
// grid (M/128), block 256.
__global__ __launch_bounds__(256) void k_gemm(
    const bf16* __restrict__ X, const bf16* __restrict__ W,
    const float* __restrict__ bias, float* __restrict__ Y, int N) {
  int wave = threadIdx.x >> 5, lane = threadIdx.x & 31;
  int r0 = blockIdx.x * 128 + wave * 16;
  v16bf xa[4];
#pragma unroll
  for (int s = 0; s < 4; ++s) xa[s] = frag_a_bf(X, kH, r0, 32 * s);
  for (int nt = 0; nt < N / 16; ++nt) {
    v8f acc = zero8();
#pragma unroll
    for (int s = 0; s < 4; ++s) {
      v16bf bw = frag_b_w(W, kH, nt * 16, 32 * s);
      acc = WMMA_BF16(xa[s], bw, acc);
    }
    int ncol = nt * 16 + (lane & 15);
    float bv = bias[ncol];
#pragma unroll
    for (int v = 0; v < 8; ++v) {
      int m = v + ((lane >> 4) << 3);
      Y[(size_t)(r0 + m) * N + ncol] = acc[v] + bv;
    }
  }
}

// ---------------- attention + out-proj (WMMA) + LayerNorm + gate + blend ----------------
// grid (B/64, C), block 256.
__global__ __launch_bounds__(256) void k_attn(
    const float* __restrict__ qkv,      // (C,B,3H)
    const float* __restrict__ hn,       // (C,B,H) f32
    const bf16* __restrict__ out_w,     // (H,H) bf16 layer slice
    const float* __restrict__ out_b,    // (H)
    const float* __restrict__ ln_g, const float* __restrict__ ln_b,  // (H)
    const float* __restrict__ gate_w,   // (2H)
    const float* __restrict__ gate_b_all, int l,
    float* __restrict__ h_out,          // d_out h_n[l] (C,B,H)
    bf16* __restrict__ x_bf) {          // (C,B,H) next-layer input
  __shared__ bf16  o_tile[64][132];
  __shared__ float proj[64][136];
  int qc = blockIdx.y;
  int b0 = blockIdx.x * 64;
  int tid = threadIdx.x;

  // phase 1: per-(row,head) attention over the 4 cells
  {
    int brow = tid >> 2, head = tid & 3;
    int b = b0 + brow;
    const float* qp = qkv + ((size_t)qc * kB + b) * kTH + head * 32;
    float q[32];
#pragma unroll
    for (int j = 0; j < 32; ++j) q[j] = qp[j];
    float sc[4];
    float mx = -1e30f;
#pragma unroll
    for (int kc = 0; kc < 4; ++kc) {
      const float* kp = qkv + ((size_t)kc * kB + b) * kTH + kH + head * 32;
      float d = 0.f;
#pragma unroll
      for (int j = 0; j < 32; ++j) d += q[j] * kp[j];
      sc[kc] = d * 0.17677669529663687f;  // 1/sqrt(32)
      mx = fmaxf(mx, sc[kc]);
    }
    float den = 0.f;
#pragma unroll
    for (int kc = 0; kc < 4; ++kc) { sc[kc] = __expf(sc[kc] - mx); den += sc[kc]; }
    float inv = 1.f / den;
    float o[32];
#pragma unroll
    for (int j = 0; j < 32; ++j) o[j] = 0.f;
#pragma unroll
    for (int kc = 0; kc < 4; ++kc) {
      const float* vp = qkv + ((size_t)kc * kB + b) * kTH + 2 * kH + head * 32;
      float a = sc[kc] * inv;
#pragma unroll
      for (int j = 0; j < 32; ++j) o[j] += a * vp[j];
    }
#pragma unroll
    for (int j = 0; j < 32; ++j) o_tile[brow][head * 32 + j] = (bf16)o[j];
  }
  __syncthreads();

  // phase 2: proj = o_tile(64x128) @ out_w^T(128x128)  via WMMA, A-frags from LDS
  int wave = tid >> 5, lane = tid & 31;
#pragma unroll
  for (int t = 0; t < 4; ++t) {
    int tt = wave * 4 + t;
    int rt = tt >> 3, ct = tt & 7;
    v8f acc = zero8();
#pragma unroll
    for (int s = 0; s < 4; ++s) {
      v16bf a;
      int row = rt * 16 + (lane & 15);
      int kb = 32 * s + ((lane >> 4) << 3);
#pragma unroll
      for (int v = 0; v < 8; ++v) {
        int k = kb + ((v >> 2) << 4) + ((v & 3) << 1);
        a[2*v]   = o_tile[row][k];
        a[2*v+1] = o_tile[row][k + 1];
      }
      v16bf bw = frag_b_w(out_w, kH, ct * 16, 32 * s);
      acc = WMMA_BF16(a, bw, acc);
    }
    int n = lane & 15;
    float ob = out_b[ct * 16 + n];
#pragma unroll
    for (int v = 0; v < 8; ++v) {
      int m = v + ((lane >> 4) << 3);
      proj[rt * 16 + m][ct * 16 + n] = acc[v] + ob;
    }
  }
  __syncthreads();

  // phase 3: LayerNorm + gate + blend (one wave handles 8 rows, shfl reductions)
  float gbias = gate_b_all[l];
  for (int rr = 0; rr < 8; ++rr) {
    int row = wave * 8 + rr;
    int b = b0 + row;
    float vals[4], hv[4];
    float s1 = 0.f, s2 = 0.f, gp = 0.f;
#pragma unroll
    for (int i = 0; i < 4; ++i) {
      int col = lane + 32 * i;
      float v = proj[row][col];
      vals[i] = v; s1 += v; s2 += v * v;
      float hh = hn[((size_t)qc * kB + b) * kH + col];
      hv[i] = hh; gp += hh * gate_w[col];
    }
#pragma unroll
    for (int m = 16; m >= 1; m >>= 1) {
      s1 += __shfl_xor(s1, m, 32);
      s2 += __shfl_xor(s2, m, 32);
      gp += __shfl_xor(gp, m, 32);
    }
    float mu = s1 * (1.f / 128.f);
    float var = s2 * (1.f / 128.f) - mu * mu;
    float rstd = rsqrtf(var + 1e-5f);
    float msg[4]; float g2 = 0.f;
#pragma unroll
    for (int i = 0; i < 4; ++i) {
      int col = lane + 32 * i;
      msg[i] = (vals[i] - mu) * rstd * ln_g[col] + ln_b[col];
      g2 += msg[i] * gate_w[kH + col];
    }
#pragma unroll
    for (int m = 16; m >= 1; m >>= 1) g2 += __shfl_xor(g2, m, 32);
    float gt = sigmoidf_(gp + g2 + gbias);
#pragma unroll
    for (int i = 0; i < 4; ++i) {
      int col = lane + 32 * i;
      float o = (1.f - gt) * hv[i] + gt * msg[i];
      size_t idx = ((size_t)qc * kB + b) * kH + col;
      h_out[idx] = o;
      x_bf[idx]  = (bf16)o;
    }
  }
}

// ---------------- host ----------------
extern "C" void kernel_launch(void* const* d_in, const int* in_sizes, int n_in,
                              void* d_out, int out_size, void* d_ws, size_t ws_size,
                              hipStream_t stream) {
  const int*   tokens   = (const int*)d_in[0];
  const float* h_in     = (const float*)d_in[1];
  const float* emb      = (const float*)d_in[2];
  const float* wih0_c0  = (const float*)d_in[3];
  const float* bih0     = (const float*)d_in[5];
  const float* whh0     = (const float*)d_in[6];
  const float* bhh0     = (const float*)d_in[7];
  const float* wih      = (const float*)d_in[8];
  const float* whh      = (const float*)d_in[9];
  const float* bih      = (const float*)d_in[10];
  const float* bhh      = (const float*)d_in[11];
  const float* ain_w    = (const float*)d_in[12];
  const float* ain_b    = (const float*)d_in[13];
  const float* aout_w   = (const float*)d_in[14];
  const float* aout_b   = (const float*)d_in[15];
  const float* ln_g     = (const float*)d_in[16];
  const float* ln_b     = (const float*)d_in[17];
  const float* gate_w   = (const float*)d_in[18];
  const float* gate_b   = (const float*)d_in[19];
  const float* head_w   = (const float*)d_in[20];
  const float* head_b   = (const float*)d_in[21];

  char* ws = (char*)d_ws;
  size_t off = 0;
  auto alloc = [&](size_t bytes) -> void* {
    void* p = ws + off;
    off = (off + bytes + 255) & ~(size_t)255;
    return p;
  };
  bf16*  x_bf   = (bf16*)alloc((size_t)kC * kB * kH * 2);
  bf16*  x0_bf  = (bf16*)alloc((size_t)kB * kE * 2);
  float* hn_f   = (float*)alloc((size_t)kC * kB * kH * 4);
  bf16*  hn_bf  = (bf16*)alloc((size_t)kC * kB * kH * 2);
  float* qkv    = (float*)alloc((size_t)kC * kB * kTH * 4);
  bf16*  w_wih0 = (bf16*)alloc((size_t)kTH * kE * 2);
  bf16*  w_wih  = (bf16*)alloc((size_t)(kL - 1) * kC * kTH * kH * 2);
  bf16*  w_whh0 = (bf16*)alloc((size_t)kC * kTH * kH * 2);
  bf16*  w_whh  = (bf16*)alloc((size_t)(kL - 1) * kC * kTH * kH * 2);
  bf16*  w_ain  = (bf16*)alloc((size_t)kL * kTH * kH * 2);
  bf16*  w_aout = (bf16*)alloc((size_t)kL * kH * kH * 2);
  bf16*  w_head = (bf16*)alloc((size_t)kO * kH * 2);
  (void)ws_size; (void)in_sizes; (void)n_in; (void)out_size;

  auto cvt = [&](const float* s, bf16* d, int n) {
    k_f2bf<<<(n + 255) / 256, 256, 0, stream>>>(s, d, n);
  };
  cvt(wih0_c0, w_wih0, kTH * kE);
  cvt(wih,     w_wih,  (kL - 1) * kC * kTH * kH);
  cvt(whh0,    w_whh0, kC * kTH * kH);
  cvt(whh,     w_whh,  (kL - 1) * kC * kTH * kH);
  cvt(ain_w,   w_ain,  kL * kTH * kH);
  cvt(aout_w,  w_aout, kL * kH * kH);
  cvt(head_w,  w_head, kO * kH);

  k_embed<<<kB, 128, 0, stream>>>(tokens, emb, x0_bf);

  float* y_out  = (float*)d_out;
  float* hn_out = (float*)d_out + (size_t)kB * kO;

  for (int l = 0; l < kL; ++l) {
    const float* h_l  = h_in + (size_t)l * kC * kB * kH;
    if (l == 0) {
      // cell 0: gi = x0 @ wih0_c0^T ; cells 1..3: gi = bias broadcast only
      k_gru<true><<<dim3(kB / 128, 1), 256, 0, stream>>>(
          x0_bf, h_l, w_wih0, w_whh0, bih0, bhh0, hn_f, hn_bf, 0);
      k_gru<false><<<dim3(kB / 128, kC - 1), 256, 0, stream>>>(
          x0_bf, h_l, w_wih0, w_whh0, bih0, bhh0, hn_f, hn_bf, 1);
    } else {
      const bf16*  Wih  = w_wih + (size_t)(l - 1) * kC * kTH * kH;
      const bf16*  Whh  = w_whh + (size_t)(l - 1) * kC * kTH * kH;
      const float* bihl = bih + (size_t)(l - 1) * kC * kTH;
      const float* bhhl = bhh + (size_t)(l - 1) * kC * kTH;
      k_gru<true><<<dim3(kB / 128, kC), 256, 0, stream>>>(
          x_bf, h_l, Wih, Whh, bihl, bhhl, hn_f, hn_bf, 0);
    }
    k_gemm<<<dim3(kC * kB / 128), 256, 0, stream>>>(
        hn_bf, w_ain + (size_t)l * kTH * kH, ain_b + (size_t)l * kTH, qkv, kTH);
    k_attn<<<dim3(kB / 64, kC), 256, 0, stream>>>(
        qkv, hn_f, w_aout + (size_t)l * kH * kH, aout_b + (size_t)l * kH,
        ln_g + (size_t)l * kH, ln_b + (size_t)l * kH, gate_w + (size_t)l * 2 * kH,
        gate_b, l, hn_out + (size_t)l * kC * kB * kH, x_bf);
  }
  // y = h_n[-1,0] @ head_w^T + head_b ; x_bf cell 0 is exactly h_n[L-1,0] in bf16
  k_gemm<<<dim3(kB / 128), 256, 0, stream>>>(x_bf, w_head, head_b, y_out, kO);
}
